// WordAttention_8744553414671
// MI455X (gfx1250) — matrix operand order repaired and checked by reading, not compiled, not run on hardware
//
#include <hip/hip_runtime.h>
#include <hip/hip_bf16.h>

#define B_   64
#define LP_  1024
#define LQ_  128
#define DIN_ 768
#define DH_  768

typedef __attribute__((ext_vector_type(16))) __bf16 v16bf;
typedef __attribute__((ext_vector_type(8)))  __bf16 bf16x8;
typedef __attribute__((ext_vector_type(8)))  float  v8f;

union Frag { v16bf v; bf16x8 h[2]; };

#define WMMA_BF16(A, Bm, C) \
  __builtin_amdgcn_wmma_f32_16x16x32_bf16(false, (A), false, (Bm), (short)0, (C), false, false)

// ---------------------------------------------------------------------------
// prep: W[d][h] f32 -> WT[h][d] bf16  (B-fragments become K-contiguous)
// ---------------------------------------------------------------------------
__global__ __launch_bounds__(256) void prep_wt(const float* __restrict__ W,
                                               __bf16* __restrict__ WT) {
  int idx = blockIdx.x * 256 + threadIdx.x;     // 768*768 threads
  int k = idx / DH_;
  int n = idx % DH_;
  WT[(size_t)n * DIN_ + k] = (__bf16)W[idx];
}

// prep: question[b][q][d] f32 -> QT[b][d][q] bf16 (B-fragments for PV GEMM)
__global__ __launch_bounds__(256) void prep_qt(const float* __restrict__ q,
                                               __bf16* __restrict__ QT) {
  size_t idx = (size_t)blockIdx.x * 256 + threadIdx.x;  // B*LQ*DIN threads
  int d = (int)(idx % DIN_);
  size_t t = idx / DIN_;
  int qq = (int)(t % LQ_);
  int b  = (int)(t / LQ_);
  QT[((size_t)b * DIN_ + d) * LQ_ + qq] = (__bf16)q[idx];
}

// prep: passage f32 -> bf16 row-major, 8 elements/thread (one-time convert so
// the 12 re-read passes of GEMM-1 move bf16 and need no v_cvt in the hot loop)
__global__ __launch_bounds__(256) void prep_p(const float* __restrict__ p,
                                              __bf16* __restrict__ PB) {
  size_t i = ((size_t)blockIdx.x * 256 + threadIdx.x) * 8;
  float4 x0 = *(const float4*)(p + i);
  float4 x1 = *(const float4*)(p + i + 4);
  bf16x8 o;
  o[0] = (__bf16)x0.x; o[1] = (__bf16)x0.y; o[2] = (__bf16)x0.z; o[3] = (__bf16)x0.w;
  o[4] = (__bf16)x1.x; o[5] = (__bf16)x1.y; o[6] = (__bf16)x1.z; o[7] = (__bf16)x1.w;
  *(bf16x8*)(PB + i) = o;
}

// ---------------------------------------------------------------------------
// WQ = relu(question @ W + bias), bf16, row-major [b][q][h]
// 1 wave = one 16-row q-tile; 8 waves / block
// ---------------------------------------------------------------------------
__global__ __launch_bounds__(256) void wq_gemm(const float*  __restrict__ q,
                                               const __bf16* __restrict__ WT,
                                               const float*  __restrict__ bias,
                                               __bf16*       __restrict__ WQ) {
  const int lane = threadIdx.x & 31;
  const int wave = threadIdx.x >> 5;
  const int lrow = lane & 15;
  const int lhi  = lane >> 4;
  const int tile = blockIdx.x * 8 + wave;       // 512 q-tiles total
  const int b    = tile >> 3;
  const int q0   = (tile & 7) * 16;

  const float* Arow = q + ((size_t)b * LQ_ + q0 + lrow) * DIN_;

  for (int n0 = 0; n0 < DH_; n0 += 32) {
    v8f acc0 = {}; v8f acc1 = {};
    for (int kk = 0; kk < DIN_; kk += 32) {
      Frag a;
      const float* pA = Arow + kk + lhi * 16;
      __builtin_prefetch(pA + 32, 0, 1);
#pragma unroll
      for (int e = 0; e < 8; ++e) a.h[0][e] = (__bf16)pA[e];
#pragma unroll
      for (int e = 0; e < 8; ++e) a.h[1][e] = (__bf16)pA[8 + e];
      Frag b0, b1;
      const __bf16* pB0 = WT + (size_t)(n0 + lrow) * DIN_ + kk + lhi * 16;
      const __bf16* pB1 = pB0 + (size_t)16 * DIN_;
      b0.h[0] = *(const bf16x8*)pB0;  b0.h[1] = *(const bf16x8*)(pB0 + 8);
      b1.h[0] = *(const bf16x8*)pB1;  b1.h[1] = *(const bf16x8*)(pB1 + 8);
      acc0 = WMMA_BF16(a.v, b0.v, acc0);
      acc1 = WMMA_BF16(a.v, b1.v, acc1);
    }
    const float bb0 = bias[n0 + lrow];
    const float bb1 = bias[n0 + 16 + lrow];
#pragma unroll
    for (int r = 0; r < 8; ++r) {
      int qrow = q0 + r + lhi * 8;
      float v0 = acc0[r] + bb0; v0 = v0 > 0.f ? v0 : 0.f;
      float v1 = acc1[r] + bb1; v1 = v1 > 0.f ? v1 : 0.f;
      __bf16* dst = WQ + ((size_t)b * LQ_ + qrow) * DH_;
      dst[n0 + lrow]      = (__bf16)v0;
      dst[n0 + 16 + lrow] = (__bf16)v1;
    }
  }
}

// ---------------------------------------------------------------------------
// Fused: Wp-tile GEMM (h-blocks of 64) -> scores -> masked softmax -> alpha @ Q
// 1 wave = 16 p-rows; block = 8 waves = 128 p-rows of one batch
// ---------------------------------------------------------------------------
#define LDS_PITCH 136   // bf16 elements; 272B rows -> conflict-friendly b128 ds ops

__global__ __launch_bounds__(256) void attn_fused(
    const __bf16*        __restrict__ PB,       // [B,1024,768] bf16 passage
    const unsigned char* __restrict__ qmask,    // [B,128] bool
    const __bf16*        __restrict__ WT,       // [768 h][768 d]
    const float*         __restrict__ bias,     // [768]
    const __bf16*        __restrict__ WQ,       // [B,128,768]
    const __bf16*        __restrict__ QT,       // [B,768,128]
    float*               __restrict__ out) {    // [B,1024,768]
  __shared__ __bf16 lds[8][16 * LDS_PITCH];     // 4352 B per wave, private region

  const int lane = threadIdx.x & 31;
  const int wave = threadIdx.x >> 5;
  const int lrow = lane & 15;
  const int lhi  = lane >> 4;
  const int b    = blockIdx.x >> 3;
  const int p0   = (blockIdx.x & 7) * 128 + wave * 16;
  __bf16* my = &lds[wave][0];

  const __bf16* Arow = PB + ((size_t)b * LP_ + p0 + lrow) * DIN_;
  const __bf16* WQb  = WQ + (size_t)b * LQ_ * DH_;

  v8f sc[8];
#pragma unroll
  for (int j = 0; j < 8; ++j) sc[j] = (v8f){};

  // ---- accumulate scores over h in blocks of 64 --------------------------
  for (int h0 = 0; h0 < DH_; h0 += 64) {
    v8f w[4];
#pragma unroll
    for (int t = 0; t < 4; ++t) w[t] = (v8f){};
    for (int kk = 0; kk < DIN_; kk += 32) {     // full-K GEMM for 16x64 Wp tile
      Frag a;
      const __bf16* pA = Arow + kk + lhi * 16;
      __builtin_prefetch(pA + 32, 0, 1);
      a.h[0] = *(const bf16x8*)pA;
      a.h[1] = *(const bf16x8*)(pA + 8);
#pragma unroll
      for (int t = 0; t < 4; ++t) {             // A reused across 4 WMMAs
        Frag bt;
        const __bf16* pW = WT + (size_t)(h0 + t * 16 + lrow) * DIN_ + kk + lhi * 16;
        bt.h[0] = *(const bf16x8*)pW;  bt.h[1] = *(const bf16x8*)(pW + 8);
        w[t] = WMMA_BF16(a.v, bt.v, w[t]);
      }
    }
    // epilogue: bias + relu, stage 16x64 bf16 tile in this wave's LDS region
#pragma unroll
    for (int t = 0; t < 4; ++t) {
      const float bb = bias[h0 + t * 16 + lrow];
#pragma unroll
      for (int r = 0; r < 8; ++r) {
        float v = w[t][r] + bb; v = v > 0.f ? v : 0.f;
        my[(r + 8 * lhi) * LDS_PITCH + t * 16 + lrow] = (__bf16)v;
      }
    }
    // reload as two A-fragments (k = h-local 0..31 and 32..63)
    Frag wa0, wa1;
    const __bf16* la = my + (size_t)lrow * LDS_PITCH + lhi * 16;
    wa0.h[0] = *(const bf16x8*)la;         wa0.h[1] = *(const bf16x8*)(la + 8);
    wa1.h[0] = *(const bf16x8*)(la + 32);  wa1.h[1] = *(const bf16x8*)(la + 40);
    // rank-64 update of the 16x128 score block (8 q-tiles)
#pragma unroll
    for (int j = 0; j < 8; ++j) {
      Frag bq0, bq1;
      const __bf16* pB = WQb + (size_t)(j * 16 + lrow) * DH_ + h0 + lhi * 16;
      bq0.h[0] = *(const bf16x8*)pB;        bq0.h[1] = *(const bf16x8*)(pB + 8);
      bq1.h[0] = *(const bf16x8*)(pB + 32); bq1.h[1] = *(const bf16x8*)(pB + 40);
      sc[j] = WMMA_BF16(wa0.v, bq0.v, sc[j]);
      sc[j] = WMMA_BF16(wa1.v, bq1.v, sc[j]);
    }
  }

  // ---- mask + softmax over q (row M spread across a 16-lane half-group) --
  const unsigned char* qm = qmask + (size_t)b * LQ_;
#pragma unroll
  for (int j = 0; j < 8; ++j) {
    if (qm[j * 16 + lrow]) {
#pragma unroll
      for (int r = 0; r < 8; ++r) sc[j][r] = -3.0e38f;
    }
  }
#pragma unroll
  for (int r = 0; r < 8; ++r) {
    float m = sc[0][r];
#pragma unroll
    for (int j = 1; j < 8; ++j) m = fmaxf(m, sc[j][r]);
#pragma unroll
    for (int off = 8; off > 0; off >>= 1) m = fmaxf(m, __shfl_xor(m, off, 32));
    float s = 0.f;
#pragma unroll
    for (int j = 0; j < 8; ++j) {
      float e = __expf(sc[j][r] - m);
      sc[j][r] = e;
      s += e;
    }
#pragma unroll
    for (int off = 8; off > 0; off >>= 1) s += __shfl_xor(s, off, 32);
    float inv = 1.0f / s;
    int mrow = r + 8 * lhi;
#pragma unroll
    for (int j = 0; j < 8; ++j)
      my[mrow * LDS_PITCH + j * 16 + lrow] = (__bf16)(sc[j][r] * inv);
  }

  // ---- out = alpha @ question  (K = 128, via QT) -------------------------
  Frag al[4];
#pragma unroll
  for (int kc = 0; kc < 4; ++kc) {
    const __bf16* la = my + (size_t)lrow * LDS_PITCH + kc * 32 + lhi * 16;
    al[kc].h[0] = *(const bf16x8*)la;
    al[kc].h[1] = *(const bf16x8*)(la + 8);
  }
  const __bf16* QTb = QT + (size_t)b * DIN_ * LQ_;
  for (int j = 0; j < 48; ++j) {
    v8f o = {};
#pragma unroll
    for (int kc = 0; kc < 4; ++kc) {
      Frag bq;
      const __bf16* pB = QTb + (size_t)(j * 16 + lrow) * LQ_ + kc * 32 + lhi * 16;
      bq.h[0] = *(const bf16x8*)pB;  bq.h[1] = *(const bf16x8*)(pB + 8);
      o = WMMA_BF16(al[kc].v, bq.v, o);
    }
#pragma unroll
    for (int r = 0; r < 8; ++r)
      out[((size_t)b * LP_ + p0 + r + 8 * lhi) * DH_ + j * 16 + lrow] = o[r];
  }
}

// ---------------------------------------------------------------------------
extern "C" void kernel_launch(void* const* d_in, const int* in_sizes, int n_in,
                              void* d_out, int out_size, void* d_ws, size_t ws_size,
                              hipStream_t stream) {
  const float*         passage  = (const float*)d_in[0];
  // d_in[1] = p_mask : unused by the reference computation
  const float*         question = (const float*)d_in[2];
  const unsigned char* q_mask   = (const unsigned char*)d_in[3];
  const float*         W_w      = (const float*)d_in[4];
  const float*         W_b      = (const float*)d_in[5];
  // d_in[6] = is_training (0 => dropout is identity)

  char* ws = (char*)d_ws;
  __bf16* WT = (__bf16*)ws;                                   // 1,179,648 B
  __bf16* WQ = (__bf16*)(ws + 1179648);                       // 12,582,912 B
  __bf16* QT = (__bf16*)(ws + 1179648 + 12582912);            // 12,582,912 B
  __bf16* PB = (__bf16*)(ws + 1179648 + 2 * 12582912);        // 100,663,296 B
  float*  outp = (float*)d_out;

  prep_wt  <<<(DIN_ * DH_) / 256,              256, 0, stream>>>(W_w, WT);
  prep_qt  <<<(B_ * LQ_ * DIN_) / 256,         256, 0, stream>>>(question, QT);
  prep_p   <<<(B_ * LP_ * DIN_) / (256 * 8),   256, 0, stream>>>(passage, PB);
  wq_gemm  <<<(B_ * LQ_) / (16 * 8),           256, 0, stream>>>(question, WT, W_b, WQ);
  attn_fused<<<B_ * (LP_ / 128),               256, 0, stream>>>(PB, q_mask, WT, W_b,
                                                                 WQ, QT, outp);
}
